// MotifGIN_39032662786191
// MI455X (gfx1250) — compile-verified, matrix operand order checked by reading, not run on hardware
//
#include <hip/hip_runtime.h>
#include <hip/hip_bf16.h>
#include <math.h>

// ---------------------------------------------------------------------------
// MotifGIN forward for MI455X (gfx1250, wave32, WMMA f16 -> f32 accumulate)
// ---------------------------------------------------------------------------

typedef __attribute__((ext_vector_type(16))) _Float16 v16h;
typedef __attribute__((ext_vector_type(8)))  _Float16 v8h;
typedef __attribute__((ext_vector_type(8)))  float    v8f;
typedef __attribute__((ext_vector_type(4)))  unsigned v4u;

union H2U { _Float16 h[2]; unsigned u; };
union V16 { v16h v; v8h h[2]; };

#define HID      256
#define N_ATOM   102400
#define M_MOTIF  16384
#define NGRAPH   256
#define EA_EDGE  409600
#define EM_EDGE  65536

#define MODE_STORE      0
#define MODE_STORE_RELU 1
#define MODE_MSG        2   // val = relu(acc + b + H[src]) ; atomicAdd agg[dst]

// ---------------------------------------------------------------------------
// Generic GEMM:  C[R x 256] = act( (A (+A2)) @ W[K x 256] + bias )
// K is a compile-time constant -> compile-time trip count, no tail guards for
// K%32==0, and pure pointer-increment staging (global loads use the 24-bit
// immediate offset field).  block = 256 threads = 8 waves; block tile =
// 16 rows x 256 cols; each wave owns two 16x16 tiles.  A/W tiles are staged
// into LDS in WMMA *fragment order* (f32->f16), consumed as two aligned 16B
// LDS reads per operand, double buffered (one barrier per K-step).
//
// CDNA5 16-bit fragment half-index: j(k) = ((k>>4)&1)*8 + (k&7),
// lane-half hi = (k>>3)&1.   R must be a multiple of 16 (true everywhere).
// ---------------------------------------------------------------------------
template<int MODE, bool HASA2, int K>
__global__ __launch_bounds__(256)
void gemm256_wmma(const float* __restrict__ A, int lda,
                  const float* __restrict__ A2,          // residual, stride HID
                  const float* __restrict__ W,           // [K x 256] row-major
                  const float* __restrict__ bias,        // [256]
                  float* __restrict__ C, int ldc,
                  const float* __restrict__ H,           // MSG: gather [*,256]
                  const int*   __restrict__ srcIdx,
                  const int*   __restrict__ dstIdx,
                  float*       __restrict__ agg)         // MSG: scatter [*,256]
{
    // fragment-order tiles, double buffered (dword-packed f16 pairs)
    __shared__ __align__(16) unsigned sA[2][32][12];    // [row+16*hi][8 dwords]+pad
    __shared__ __align__(16) unsigned sW[2][256][20];   // [col][hi*8 + 8 dwords]+pad

    const int tid  = threadIdx.x;
    const int wave = tid >> 5;
    const int lane = tid & 31;
    const int hi   = lane >> 4;
    const int lo   = lane & 15;
    const int row0 = blockIdx.x * 16;
    const int colw = wave * 32;

    constexpr int KSTEPS = (K + 31) >> 5;

    // A staging: thread (r = tid>>4, q = tid&15) packs two consecutive k.
    const int ar  = tid >> 4;            // tile row
    const int aq  = tid & 15;
    const int ahi = aq >> 3;             // which lane-half this data feeds
    const int aj  = (aq & 7) * 2;        // half-slot
    const int akl = ((aj >> 3) & 1) * 16 + ahi * 8 + (aj & 7);

    const float* aPtr  = A + (long)(row0 + ar) * lda + akl;
    const float* a2Ptr = HASA2 ? (A2 + (long)(row0 + ar) * HID + akl) : nullptr;
    const float* wPtr  = W + tid;        // thread owns output column c = tid

    auto stageA = [&](int s, int buf) {
        float v0, v1;
        if ((K & 31) == 0) {
            v0 = aPtr[0];
            v1 = aPtr[1];
            if (HASA2) { v0 += a2Ptr[0]; v1 += a2Ptr[1]; }
        } else {                         // only the K=16 instantiations
            int gk = (s << 5) + akl;
            int i0 = (gk     < K) ? akl     : 0;
            int i1 = (gk + 1 < K) ? akl + 1 : 0;
            v0 = (gk     < K) ? aPtr[i0 - akl] : 0.0f;
            v1 = (gk + 1 < K) ? aPtr[i1 - akl] : 0.0f;
        }
        H2U p; p.h[0] = (_Float16)v0; p.h[1] = (_Float16)v1;
        sA[buf][ar + 16 * ahi][aj >> 1] = p.u;
        aPtr += 32;
        if (HASA2) a2Ptr += 32;
    };

    auto stageW = [&](int s, int buf) {
        unsigned packs[16];
        #pragma unroll
        for (int q = 0; q < 16; ++q) {
            const int k = ((q >> 2) & 1) * 16 + ((q >> 3) & 1) * 8 + (q & 3) * 2;
            float v0, v1;
            if ((K & 31) == 0) {
                v0 = wPtr[(long)k * HID];          // immediate-offset loads
                v1 = wPtr[(long)(k + 1) * HID];
            } else {
                int gk = (s << 5) + k;
                v0 = (gk     < K) ? wPtr[(long)k * HID]       : 0.0f;
                v1 = (gk + 1 < K) ? wPtr[(long)(k + 1) * HID] : 0.0f;
            }
            H2U p; p.h[0] = (_Float16)v0; p.h[1] = (_Float16)v1;
            packs[q] = p.u;
        }
        #pragma unroll
        for (int q = 0; q < 4; ++q) {
            v4u t;
            t.x = packs[4 * q];     t.y = packs[4 * q + 1];
            t.z = packs[4 * q + 2]; t.w = packs[4 * q + 3];
            *(v4u*)&sW[buf][tid][4 * q] = t;
        }
        wPtr += 32 * HID;
    };

    v8f acc0 = {}, acc1 = {};
    stageA(0, 0);
    stageW(0, 0);
    __syncthreads();

    for (int s = 0; s < KSTEPS; ++s) {
        const int buf = s & 1;
        if (s + 1 < KSTEPS) { stageA(s + 1, buf ^ 1); stageW(s + 1, buf ^ 1); }
        V16 a, b0, b1;
        a.h[0]  = *(const v8h*)&sA[buf][lane][0];
        a.h[1]  = *(const v8h*)&sA[buf][lane][4];
        b0.h[0] = *(const v8h*)&sW[buf][colw + lo][hi * 8];
        b0.h[1] = *(const v8h*)&sW[buf][colw + lo][hi * 8 + 4];
        b1.h[0] = *(const v8h*)&sW[buf][colw + 16 + lo][hi * 8];
        b1.h[1] = *(const v8h*)&sW[buf][colw + 16 + lo][hi * 8 + 4];
        acc0 = __builtin_amdgcn_wmma_f32_16x16x32_f16(false, a.v, false, b0.v,
                                                      (short)0, acc0, false, false);
        acc1 = __builtin_amdgcn_wmma_f32_16x16x32_f16(false, a.v, false, b1.v,
                                                      (short)0, acc1, false, false);
        __syncthreads();
    }

    // epilogue (C layout: vgpr v -> row v + 8*hi ; col = lane&15)
    const int col0 = colw + lo;
    const int col1 = col0 + 16;
    const float bb0 = bias[col0];
    const float bb1 = bias[col1];
    #pragma unroll
    for (int v = 0; v < 8; ++v) {
        const int gr = row0 + v + 8 * hi;     // always < R (R % 16 == 0)
        float v0 = acc0[v] + bb0;
        float v1 = acc1[v] + bb1;
        if (MODE == MODE_MSG) {
            int si = srcIdx[gr];
            int di = dstIdx[gr];
            v0 = fmaxf(v0 + H[(long)si * HID + col0], 0.0f);
            v1 = fmaxf(v1 + H[(long)si * HID + col1], 0.0f);
            atomicAdd(&agg[(long)di * HID + col0], v0);
            atomicAdd(&agg[(long)di * HID + col1], v1);
        } else {
            if (MODE == MODE_STORE_RELU) { v0 = fmaxf(v0, 0.0f); v1 = fmaxf(v1, 0.0f); }
            C[(long)gr * ldc + col0] = v0;
            C[(long)gr * ldc + col1] = v1;
        }
    }
}

// ---------------------------------------------------------------------------
// alpha[r] = sigmoid( dot(Hm[r], w) )   (one wave32 per row)
// ---------------------------------------------------------------------------
__global__ __launch_bounds__(256)
void rowdot_sigmoid(const float* __restrict__ Hm, const float* __restrict__ w,
                    float* __restrict__ out, int R)
{
    int wid  = (int)((blockIdx.x * blockDim.x + threadIdx.x) >> 5);
    int lane = threadIdx.x & 31;
    if (wid >= R) return;
    const float* row = Hm + (long)wid * HID;
    float s = 0.0f;
    #pragma unroll
    for (int i = 0; i < 8; ++i) s += row[lane + 32 * i] * w[lane + 32 * i];
    #pragma unroll
    for (int off = 16; off; off >>= 1) s += __shfl_xor(s, off, 32);
    if (lane == 0) out[wid] = 1.0f / (1.0f + __expf(-s));
}

// ---------------------------------------------------------------------------
// pair_alpha[e] = sigmoid(h[aIdx[e]] . w_pair);  hMotif[mIdx[e]] += pa*h[aIdx[e]]
// ---------------------------------------------------------------------------
__global__ __launch_bounds__(256)
void motif_pool(const float* __restrict__ Hm, const float* __restrict__ wpair,
                const int* __restrict__ mIdx, const int* __restrict__ aIdx,
                float* __restrict__ pairAlpha, float* __restrict__ hMotif, int R)
{
    int wid  = (int)((blockIdx.x * blockDim.x + threadIdx.x) >> 5);
    int lane = threadIdx.x & 31;
    if (wid >= R) return;
    int arow = aIdx[wid];
    const float* row = Hm + (long)arow * HID;
    float s = 0.0f;
    #pragma unroll
    for (int i = 0; i < 8; ++i) s += row[lane + 32 * i] * wpair[lane + 32 * i];
    #pragma unroll
    for (int off = 16; off; off >>= 1) s += __shfl_xor(s, off, 32);
    float pa = 1.0f / (1.0f + __expf(-s));
    if (lane == 0) pairAlpha[wid] = pa;
    float* d = hMotif + (long)mIdx[wid] * HID;
    #pragma unroll
    for (int i = 0; i < 8; ++i) atomicAdd(&d[lane + 32 * i], pa * row[lane + 32 * i]);
}

// Xcat[m] = [ type_emb[motif_type[m]] (64) | h_motif[m] (256) ]   -> [M,320]
__global__ void build_xcat(const float* __restrict__ typeEmb,
                           const int*   __restrict__ motifType,
                           const float* __restrict__ hMotif,
                           float* __restrict__ Xcat)
{
    int m = blockIdx.x, t = threadIdx.x;   // 320 threads
    float v = (t < 64) ? typeEmb[motifType[m] * 64 + t]
                       : hMotif[(long)m * HID + (t - 64)];
    Xcat[(long)m * 320 + t] = v;
}

// X2[e] cols 0..63 = couple ; cols 320..575 = node_emb  (cols 64..319 via GEMM)
__global__ void build_xcat2(const float* __restrict__ mea,        // [EM,20]
                            const int*   __restrict__ meSrc,      // motif_edge_index[0]
                            const int*   __restrict__ motifBatch,
                            const int*   __restrict__ atomPtr,
                            const float* __restrict__ typeEmb,
                            const float* __restrict__ xAtom,      // [N,256]
                            float* __restrict__ X2)               // [EM,576]
{
    int e = blockIdx.x, t = threadIdx.x;   // 320 threads
    const float* row = mea + (long)e * 20;
    if (t < 64) {
        int st = (int)row[0], dt = (int)row[1];
        X2[(long)e * 576 + t] = typeEmb[st * 64 + t] + typeEmb[dt * 64 + t];
    } else {
        int g   = motifBatch[meSrc[e]];
        int off = atomPtr[g];
        int gs  = (int)row[19] + off;  // local_atom_src -> global
        int gd  = (int)row[18] + off;  // local_atom_dst -> global
        int c   = t - 64;
        X2[(long)e * 576 + 320 + c] =
            xAtom[(long)gs * HID + c] + xAtom[(long)gd * HID + c];
    }
}

// motif_level[g] = sum of 64 consecutive hm rows (motif_batch = repeat(arange, 64))
__global__ void graph_pool(const float* __restrict__ hm, float* __restrict__ out)
{
    int g = blockIdx.x, c = threadIdx.x;   // 256 threads
    const float* base = hm + (long)g * 64 * HID + c;
    float s = 0.0f;
    #pragma unroll 8
    for (int i = 0; i < 64; ++i) s += base[(long)i * HID];
    out[(long)g * HID + c] = s;
}

__global__ void zero_f32(float* __restrict__ p, long n)
{
    long i      = (long)blockIdx.x * blockDim.x + threadIdx.x;
    long stride = (long)gridDim.x * blockDim.x;
    for (; i < n; i += stride) p[i] = 0.0f;
}

// ---------------------------------------------------------------------------
extern "C" void kernel_launch(void* const* d_in, const int* in_sizes, int n_in,
                              void* d_out, int out_size, void* d_ws, size_t ws_size,
                              hipStream_t stream)
{
    (void)in_sizes; (void)n_in; (void)out_size; (void)ws_size;
    const long Nl = N_ATOM, Ml = M_MOTIF;

    // inputs (top-level dict order; nested params flattened by sorted keys, W before b)
    const float* atom_x         = (const float*)d_in[0];
    const float* atom_edge_attr = (const float*)d_in[1];
    const float* motif_edge_att = (const float*)d_in[2];
    const float* W_atom = (const float*)d_in[3];  const float* b_atom = (const float*)d_in[4];
    const float* W_eenc = (const float*)d_in[5];  const float* b_eenc = (const float*)d_in[6];
    const float* W_enn  = (const float*)d_in[7];  const float* b_enn  = (const float*)d_in[8];
    const float* gW[2][3]; const float* gB[2][3];
    for (int l = 0; l < 2; ++l)
        for (int k = 0; k < 3; ++k) {
            gW[l][k] = (const float*)d_in[9 + l * 6 + k * 2];
            gB[l][k] = (const float*)d_in[10 + l * 6 + k * 2];
        }
    const float* mW[2][3]; const float* mB[2][3];
    for (int l = 0; l < 2; ++l)
        for (int k = 0; k < 3; ++k) {
            mW[l][k] = (const float*)d_in[21 + l * 6 + k * 2];
            mB[l][k] = (const float*)d_in[22 + l * 6 + k * 2];
        }
    const float* W_nnn   = (const float*)d_in[33]; const float* b_nnn = (const float*)d_in[34];
    const float* typeEmb = (const float*)d_in[35];
    const float* w_node  = (const float*)d_in[36];
    const float* w_pair  = (const float*)d_in[37];
    const int* atom_ptr    = (const int*)d_in[38];
    const int* motif_type  = (const int*)d_in[39];
    const int* motif_batch = (const int*)d_in[40];
    const int* a_src = (const int*)d_in[41];           // atom_edge_index
    const int* a_dst = a_src + EA_EDGE;
    const int* c_m   = (const int*)d_in[42];           // contains_edge_index
    const int* c_a   = c_m + N_ATOM;
    const int* m_src = (const int*)d_in[43];           // motif_edge_index
    const int* m_dst = m_src + EM_EDGE;

    // output layout: node_alpha | pair_alpha | hm | x_atom | motif_level
    float* out  = (float*)d_out;
    float* o_na = out;
    float* o_pa = out + Nl;
    float* o_hm = out + 2 * Nl;
    float* o_xa = o_hm + Ml * HID;
    float* o_ml = o_xa + Nl * HID;

    // workspace
    float* ws     = (float*)d_ws;
    float* ea     = ws;                         // [EA,256]
    float* h      = ea + (long)EA_EDGE * HID;   // [N,256]
    float* aggA   = h + Nl * HID;               // [N,256]
    float* u      = aggA + Nl * HID;            // [N,256]
    float* hMotif = u + Nl * HID;               // [M,256]
    float* Xcat   = hMotif + Ml * HID;          // [M,320]
    float* xh     = Xcat + Ml * 320;            // [M,256]
    float* X2     = ea;                         // alias: [EM,576] (after atom GNN)
    float* h_me   = ea + (long)EM_EDGE * 576;   // alias inside ea region: [EM,256]
    float* aggM   = aggA;                       // alias [M,256]
    float* uM     = u;                          // alias [M,256]

    dim3 blk(256);

    // 1) x_atom = relu(atom_x @ W_atom + b) -> straight into d_out
    gemm256_wmma<MODE_STORE_RELU, false, 128><<<dim3(N_ATOM / 16), blk, 0, stream>>>(
        atom_x, 128, nullptr, W_atom, b_atom, o_xa, HID,
        nullptr, nullptr, nullptr, nullptr);

    // 2) ea = relu(atom_edge_attr @ W_eenc + b)
    gemm256_wmma<MODE_STORE_RELU, false, 16><<<dim3(EA_EDGE / 16), blk, 0, stream>>>(
        atom_edge_attr, 16, nullptr, W_eenc, b_eenc, ea, HID,
        nullptr, nullptr, nullptr, nullptr);

    // 3) atom GINE layers
    const float* hcur = o_xa;
    for (int l = 0; l < 2; ++l) {
        zero_f32<<<dim3(2048), blk, 0, stream>>>(aggA, Nl * HID);
        // fused: agg[dst] += relu(h[src] + ea@W_edge + b)
        gemm256_wmma<MODE_MSG, false, 256><<<dim3(EA_EDGE / 16), blk, 0, stream>>>(
            ea, HID, nullptr, gW[l][0], gB[l][0], nullptr, 0,
            hcur, a_src, a_dst, aggA);
        // u = relu((h + agg) @ mlp1 + b1)
        gemm256_wmma<MODE_STORE_RELU, true, 256><<<dim3(N_ATOM / 16), blk, 0, stream>>>(
            hcur, HID, aggA, gW[l][1], gB[l][1], u, HID,
            nullptr, nullptr, nullptr, nullptr);
        // h = u @ mlp2 + b2
        gemm256_wmma<MODE_STORE, false, 256><<<dim3(N_ATOM / 16), blk, 0, stream>>>(
            u, HID, nullptr, gW[l][2], gB[l][2], h, HID,
            nullptr, nullptr, nullptr, nullptr);
        hcur = h;
    }

    // 4) node_alpha, pair_alpha + attentive motif pooling
    rowdot_sigmoid<<<dim3(N_ATOM / 8), blk, 0, stream>>>(h, w_node, o_na, N_ATOM);
    zero_f32<<<dim3(1024), blk, 0, stream>>>(hMotif, Ml * HID);
    motif_pool<<<dim3(N_ATOM / 8), blk, 0, stream>>>(
        h, w_pair, c_m, c_a, o_pa, hMotif, N_ATOM);

    // 5) h_motif_atom = node_nn(concat(type_emb, h_motif))
    build_xcat<<<dim3(M_MOTIF), dim3(320), 0, stream>>>(typeEmb, motif_type, hMotif, Xcat);
    gemm256_wmma<MODE_STORE, false, 320><<<dim3(M_MOTIF / 16), blk, 0, stream>>>(
        Xcat, 320, nullptr, W_nnn, b_nnn, xh, HID,
        nullptr, nullptr, nullptr, nullptr);

    // 6) motif-edge embedding: X2 = [couple | relu(edge_enc(ef)) | node_emb]
    build_xcat2<<<dim3(EM_EDGE), dim3(320), 0, stream>>>(
        motif_edge_att, m_src, motif_batch, atom_ptr, typeEmb, o_xa, X2);
    gemm256_wmma<MODE_STORE_RELU, false, 16><<<dim3(EM_EDGE / 16), blk, 0, stream>>>(
        motif_edge_att + 2, 20, nullptr, W_eenc, b_eenc, X2 + 64, 576,
        nullptr, nullptr, nullptr, nullptr);
    gemm256_wmma<MODE_STORE, false, 576><<<dim3(EM_EDGE / 16), blk, 0, stream>>>(
        X2, 576, nullptr, W_enn, b_enn, h_me, HID,
        nullptr, nullptr, nullptr, nullptr);

    // 7) motif GINE layers (mp1, mp2); final hm -> d_out
    const float* xcur = xh;
    for (int l = 0; l < 2; ++l) {
        zero_f32<<<dim3(1024), blk, 0, stream>>>(aggM, Ml * HID);
        gemm256_wmma<MODE_MSG, false, 256><<<dim3(EM_EDGE / 16), blk, 0, stream>>>(
            h_me, HID, nullptr, mW[l][0], mB[l][0], nullptr, 0,
            xcur, m_src, m_dst, aggM);
        gemm256_wmma<MODE_STORE_RELU, true, 256><<<dim3(M_MOTIF / 16), blk, 0, stream>>>(
            xcur, HID, aggM, mW[l][1], mB[l][1], uM, HID,
            nullptr, nullptr, nullptr, nullptr);
        float* dst = (l == 0) ? xh : o_hm;
        gemm256_wmma<MODE_STORE, false, 256><<<dim3(M_MOTIF / 16), blk, 0, stream>>>(
            uM, HID, nullptr, mW[l][2], mB[l][2], dst, HID,
            nullptr, nullptr, nullptr, nullptr);
        xcur = dst;
    }

    // 8) motif_level = global_add_pool(hm)  (64 contiguous rows per graph)
    graph_pool<<<dim3(NGRAPH), blk, 0, stream>>>(o_hm, o_ml);
}
